// NSM_71854802862763
// MI455X (gfx1250) — matrix operand order repaired and checked by reading, not compiled, not run on hardware
//
#include <hip/hip_runtime.h>
#include <math.h>

typedef __attribute__((ext_vector_type(16))) __bf16 v16bf;
typedef __attribute__((ext_vector_type(8)))  __bf16 v8bf;
typedef __attribute__((ext_vector_type(8)))  float  v8f;

#define Bq 64
#define Lq 20
#define Hq 300
#define Pq 8
#define Nn 32768
#define Ee 262144
#define Cv 2048
#define OUTq 1845
#define NSTEPS 3
#define NTILES 19      // ceil(300/16) real tiles
#define NT_PAD 20      // padded so waves 0..7 take {w, w+8} and waves 0..3 take {w+16}
#define KT_PER_H 10    // ceil(300/32)

__device__ __forceinline__ float eluf(float x){ return x > 0.f ? x : (__expf(x) - 1.f); }
__device__ __forceinline__ float sigf(float x){ return 1.f / (1.f + __expf(-x)); }

__device__ __forceinline__ unsigned short bfbits(float f){
    unsigned u = __float_as_uint(f);
    return (unsigned short)((u + 0x7fffu + ((u >> 16) & 1u)) >> 16);   // RNE
}
__device__ __forceinline__ __bf16 to_bf16(float f){
    unsigned short s = bfbits(f);
    __bf16 b;
    __builtin_memcpy(&b, &s, 2);
    return b;
}
#define SHUF16(lo, hi) __builtin_shufflevector((lo), (hi), 0,1,2,3,4,5,6,7,8,9,10,11,12,13,14,15)

// ---------------------------------------------------------------- utilities
__global__ void k_zero(float* p, int n){
    int i = blockIdx.x * blockDim.x + threadIdx.x;
    if (i < n) p[i] = 0.f;
}

// ---------------------------------------------------------------- weight pre-pack (once per launch)
// Pack We (H x H) into per-lane WMMA B-fragment layout (NT_PAD tiles, tail tile zero):
// idx = (((kt*NT_PAD + nt)*32 + lane) << 4) + e ; element e holds K = kt*32 + 8*half + e + (e >= 8 ? 8 : 0)
__global__ void k_pack_we(const float* __restrict__ We, __bf16* __restrict__ out){
    int i = blockIdx.x * blockDim.x + threadIdx.x;
    if (i >= KT_PER_H * NT_PAD * 32 * 16) return;
    int e    = i & 15;
    int lane = (i >> 4) & 31;
    int rest = i >> 9;
    int nt = rest % NT_PAD, kt = rest / NT_PAD;
    int half = lane >> 4, ln = lane & 15;
    int kk = kt * 32 + 8 * half + e + ((e >> 3) << 3);
    int n  = nt * 16 + ln;
    float v = (kk < Hq && n < Hq) ? We[(size_t)kk * Hq + n] : 0.f;
    out[i] = to_bf16(v);
}

// Pack Wp (P x H x H): idx = ((((p*KT+kt)*NT_PAD + nt)*32 + lane) << 4) + e
__global__ void k_pack_wp(const float* __restrict__ Wp, __bf16* __restrict__ out){
    int i = blockIdx.x * blockDim.x + threadIdx.x;
    if (i >= Pq * KT_PER_H * NT_PAD * 32 * 16) return;
    int e    = i & 15;
    int lane = (i >> 4) & 31;
    int rest = i >> 9;
    int nt = rest % NT_PAD;
    int r2 = rest / NT_PAD;
    int kt = r2 % KT_PER_H, p = r2 / KT_PER_H;
    int half = lane >> 4, ln = lane & 15;
    int kk = kt * 32 + 8 * half + e + ((e >> 3) << 3);
    int n  = nt * 16 + ln;
    float v = (kk < Hq && n < Hq) ? Wp[(size_t)p * Hq * Hq + (size_t)kk * Hq + n] : 0.f;
    out[i] = to_bf16(v);
}

// ---------------------------------------------------------------- tagging
__global__ void k_qk(const float* __restrict__ q, const float* __restrict__ wtag,
                     float* __restrict__ qk){
    int i = blockIdx.x * blockDim.x + threadIdx.x;
    if (i >= Bq * Lq * Hq) return;
    int bl = i / Hq, k = i % Hq;
    const float* qr = q + (size_t)bl * Hq;
    float s = 0.f;
    for (int h = 0; h < Hq; ++h) s += qr[h] * wtag[h * Hq + k];
    qk[i] = s;
}

__global__ void k_logits(const float* __restrict__ qk, const float* __restrict__ cv,
                         const float* __restrict__ defe, float* __restrict__ logit){
    long i = (long)blockIdx.x * blockDim.x + threadIdx.x;
    if (i >= (long)Bq * Lq * (Cv + 1)) return;
    int bl = (int)(i / (Cv + 1)), c = (int)(i % (Cv + 1));
    const float* row = (c < Cv) ? (cv + (size_t)c * Hq) : defe;
    const float* qr  = qk + (size_t)bl * Hq;
    float s = 0.f;
    for (int k = 0; k < Hq; ++k) s += qr[k] * row[k];
    logit[i] = s;
}

__global__ void k_softmax_rows(float* __restrict__ x, int rows, int R){
    int r = blockIdx.x;
    if (r >= rows) return;
    float* p = x + (size_t)r * R;
    __shared__ float red[256];
    int tid = threadIdx.x;
    float m = -1e30f;
    for (int j = tid; j < R; j += 256) m = fmaxf(m, p[j]);
    red[tid] = m; __syncthreads();
    for (int s = 128; s > 0; s >>= 1){ if (tid < s) red[tid] = fmaxf(red[tid], red[tid + s]); __syncthreads(); }
    m = red[0]; __syncthreads();
    float sum = 0.f;
    for (int j = tid; j < R; j += 256){ float e = __expf(p[j] - m); p[j] = e; sum += e; }
    red[tid] = sum; __syncthreads();
    for (int s = 128; s > 0; s >>= 1){ if (tid < s) red[tid] += red[tid + s]; __syncthreads(); }
    float inv = 1.f / red[0];
    for (int j = tid; j < R; j += 256) p[j] *= inv;
}

__global__ void k_tagged(const float* __restrict__ sim, const float* __restrict__ q,
                         const float* __restrict__ cv, float* __restrict__ tagged){
    int i = blockIdx.x * blockDim.x + threadIdx.x;
    if (i >= Bq * Lq * Hq) return;
    int bl = i / Hq, h = i % Hq;
    const float* s = sim + (size_t)bl * (Cv + 1);
    float acc = s[Cv] * q[i];
    for (int c = 0; c < Cv; ++c) acc += s[c] * cv[(size_t)c * Hq + h];
    tagged[i] = acc;
}

// ---------------------------------------------------------------- LSTM encoder
__global__ void k_lstm_z(const float* __restrict__ tagged, int t, const float* __restrict__ h,
                         const float* __restrict__ Wih, const float* __restrict__ Whh,
                         const float* __restrict__ bih, const float* __restrict__ bhh,
                         float* __restrict__ z){
    int i = blockIdx.x * blockDim.x + threadIdx.x;
    if (i >= Bq * 4 * Hq) return;
    int b = i / (4 * Hq), j = i % (4 * Hq);
    const float* x  = tagged + ((size_t)b * Lq + t) * Hq;
    const float* hr = h + (size_t)b * Hq;
    const float* wi = Wih + (size_t)j * Hq;
    const float* wh = Whh + (size_t)j * Hq;
    float s = bih[j] + bhh[j];
    for (int k = 0; k < Hq; ++k) s += x[k] * wi[k] + hr[k] * wh[k];
    z[i] = s;
}

__global__ void k_lstm_upd(const float* __restrict__ z, float* __restrict__ h, float* __restrict__ c){
    int i = blockIdx.x * blockDim.x + threadIdx.x;
    if (i >= Bq * Hq) return;
    int b = i / Hq, u = i % Hq;
    const float* zb = z + (size_t)b * 4 * Hq;
    float ig = sigf(zb[u]);
    float fg = sigf(zb[Hq + u]);
    float gg = tanhf(zb[2 * Hq + u]);
    float og = sigf(zb[3 * Hq + u]);
    float cc = fg * c[i] + ig * gg;
    c[i] = cc;
    h[i] = og * tanhf(cc);
}

// ---------------------------------------------------------------- RNN instruction decoder
__global__ void k_rnn(const float* __restrict__ enc, const float* __restrict__ hxin,
                      const float* __restrict__ Wih, const float* __restrict__ Whh,
                      const float* __restrict__ bih, const float* __restrict__ bhh,
                      float* __restrict__ hxout, float* __restrict__ hidden, int s){
    int i = blockIdx.x * blockDim.x + threadIdx.x;
    if (i >= Bq * Hq) return;
    int b = i / Hq, u = i % Hq;
    const float* e  = enc  + (size_t)b * Hq;
    const float* hh = hxin + (size_t)b * Hq;
    const float* wi = Wih + (size_t)u * Hq;
    const float* wh = Whh + (size_t)u * Hq;
    float acc = bih[u] + bhh[u];
    for (int k = 0; k < Hq; ++k) acc += e[k] * wi[k] + hh[k] * wh[k];
    acc = fmaxf(acc, 0.f);
    hxout[i] = acc;
    hidden[((size_t)b * (NSTEPS + 1) + s) * Hq + u] = acc;
}

__global__ void k_instr(const float* __restrict__ hidden, const float* __restrict__ tagged,
                        float* __restrict__ instructions){
    int bi = blockIdx.x;
    int b = bi / (NSTEPS + 1), ii = bi % (NSTEPS + 1);
    __shared__ float logit[Lq];
    __shared__ float a[Lq];
    int tid = threadIdx.x;
    const float* hv = hidden + ((size_t)b * (NSTEPS + 1) + ii) * Hq;
    if (tid < Lq){
        const float* tg = tagged + ((size_t)b * Lq + tid) * Hq;
        float s = 0.f;
        for (int k = 0; k < Hq; ++k) s += hv[k] * tg[k];
        logit[tid] = s;
    }
    __syncthreads();
    if (tid == 0){
        float m = -1e30f;
        for (int l = 0; l < Lq; ++l) m = fmaxf(m, logit[l]);
        float sum = 0.f;
        for (int l = 0; l < Lq; ++l){ float e = __expf(logit[l] - m); a[l] = e; sum += e; }
        float inv = 1.f / sum;
        for (int l = 0; l < Lq; ++l) a[l] *= inv;
    }
    __syncthreads();
    for (int h = tid; h < Hq; h += blockDim.x){
        float s = 0.f;
        for (int l = 0; l < Lq; ++l) s += a[l] * tagged[((size_t)b * Lq + l) * Hq + h];
        instructions[((size_t)b * (NSTEPS + 1) + ii) * Hq + h] = s;
    }
}

// ---------------------------------------------------------------- graph setup
__global__ void k_counts(const int* __restrict__ nidx, float* __restrict__ counts){
    int n = blockIdx.x * blockDim.x + threadIdx.x;
    if (n < Nn) atomicAdd(&counts[nidx[n]], 1.f);
}
__global__ void k_dist_init(const int* __restrict__ nidx, const float* __restrict__ counts,
                            float* __restrict__ dist){
    int n = blockIdx.x * blockDim.x + threadIdx.x;
    if (n < Nn) dist[n] = 1.f / counts[nidx[n]];
}

__global__ void k_psim(const float* __restrict__ instructions, int step,
                       const float* __restrict__ prop, float* __restrict__ psim){
    int b = blockIdx.x, tid = threadIdx.x;
    __shared__ float d[Pq + 1];
    const float* iv = instructions + ((size_t)b * (NSTEPS + 1) + step) * Hq;
    if (tid <= Pq){
        const float* pr = prop + (size_t)tid * Hq;
        float s = 0.f;
        for (int k = 0; k < Hq; ++k) s += iv[k] * pr[k];
        d[tid] = s;
    }
    __syncthreads();
    if (tid == 0){
        float m = -1e30f;
        for (int j = 0; j <= Pq; ++j) m = fmaxf(m, d[j]);
        float e[Pq + 1]; float sum = 0.f;
        for (int j = 0; j <= Pq; ++j){ e[j] = __expf(d[j] - m); sum += e[j]; }
        float inv = 1.f / sum;
        for (int j = 0; j <= Pq; ++j) psim[b * (Pq + 1) + j] = e[j] * inv;
    }
}

// ---------------------------------------------------------------- epilogue helper
__device__ __forceinline__ float frag_dot(const v8f& acc, int r_unused){ return 0.f; }

// ---------------------------------------------------------------- node WMMA kernel
// ns_raw[n] = sum_k elu( sum_{p,h} npsim[b,p]*instr[b,h]*node_attrs[n,p,h]*Wp[p,h,k] ) * w_ns[k]
__global__ __launch_bounds__(256) void k_node_scores(
        const float* __restrict__ node_attrs, const float* __restrict__ instructions, int step,
        const float* __restrict__ psim, const __bf16* __restrict__ wppk,
        const int* __restrict__ nidx, const float* __restrict__ w_ns, float* __restrict__ ns_dot){
    __shared__ __bf16 As[16][320];
    __shared__ int bsh[16];
    int tid = threadIdx.x, lane = tid & 31;
    int wavU = __builtin_amdgcn_readfirstlane(tid >> 5);   // uniform (SGPR) wave id
    int r0 = blockIdx.x * 16;
    if (tid < 16){
        bsh[tid] = nidx[r0 + tid];
        if (r0 + 16 + tid < Nn)
            __builtin_prefetch(node_attrs + (size_t)(r0 + 16 + tid) * Pq * Hq, 0, 1);
    }
    __syncthreads();
    int half = lane >> 4, ln = lane & 15;
    int rr = tid >> 4, kb = (tid & 15) * 2;       // staging: row, even-k start

    v8f acc0, acc1, acc2;
#pragma unroll
    for (int r = 0; r < 8; ++r){ acc0[r] = 0.f; acc1[r] = 0.f; acc2[r] = 0.f; }

    for (int p = 0; p < Pq; ++p){
        {   // stage scaled A tile for property p (f32 -> bf16 pairs, b32 LDS stores)
            int b = bsh[rr];
            float sc = psim[b * (Pq + 1) + p];
            const float* xb  = instructions + ((size_t)b * (NSTEPS + 1) + step) * Hq;
            const float* nab = node_attrs + ((size_t)(r0 + rr) * Pq + p) * Hq;
            for (int k0 = kb; k0 < 320; k0 += 32){
                float v0 = (k0     < Hq) ? sc * xb[k0]     * nab[k0]     : 0.f;
                float v1 = (k0 + 1 < Hq) ? sc * xb[k0 + 1] * nab[k0 + 1] : 0.f;
                *(unsigned*)&As[rr][k0] = (unsigned)bfbits(v0) | ((unsigned)bfbits(v1) << 16);
            }
        }
        __syncthreads();
        for (int kt = 0; kt < KT_PER_H; ++kt){
            const v8bf* pa = (const v8bf*)&As[ln][(kt << 5) + (half << 3)];
            v16bf a = SHUF16(pa[0], pa[2]);       // 2x ds_load_b128 per fragment
            const __bf16* wb = wppk + (((size_t)(p * KT_PER_H + kt) * NT_PAD) << 9) + ((size_t)lane << 4);
            {   // nt = wavU
                const v8bf* pb = (const v8bf*)(wb + ((size_t)wavU << 9));
                v16bf bf = SHUF16(pb[0], pb[1]);
                acc0 = __builtin_amdgcn_wmma_f32_16x16x32_bf16(false, a, false, bf, (short)0, acc0, false, false);
            }
            {   // nt = wavU + 8
                const v8bf* pb = (const v8bf*)(wb + ((size_t)(wavU + 8) << 9));
                v16bf bf = SHUF16(pb[0], pb[1]);
                acc1 = __builtin_amdgcn_wmma_f32_16x16x32_bf16(false, a, false, bf, (short)0, acc1, false, false);
            }
            if (wavU < 4){   // nt = wavU + 16 (tile 19 zero-padded): scalar branch, EXEC untouched
                const v8bf* pb = (const v8bf*)(wb + ((size_t)(wavU + 16) << 9));
                v16bf bf = SHUF16(pb[0], pb[1]);
                acc2 = __builtin_amdgcn_wmma_f32_16x16x32_bf16(false, a, false, bf, (short)0, acc2, false, false);
            }
        }
        __syncthreads();
    }
    // fused epilogue: elu -> dot(w_ns) -> lane butterfly reduce -> per-node accumulate
    {
        int col = wavU * 16 + ln;
        float wv = (col < Hq) ? w_ns[col] : 0.f;
#pragma unroll
        for (int r = 0; r < 8; ++r){
            float v = eluf(acc0[r]) * wv;
            for (int m = 1; m < 16; m <<= 1) v += __shfl_xor(v, m, 32);
            if (ln == 0) atomicAdd(&ns_dot[r0 + r + 8 * half], v);
        }
    }
    {
        int col = (wavU + 8) * 16 + ln;
        float wv = (col < Hq) ? w_ns[col] : 0.f;
#pragma unroll
        for (int r = 0; r < 8; ++r){
            float v = eluf(acc1[r]) * wv;
            for (int m = 1; m < 16; m <<= 1) v += __shfl_xor(v, m, 32);
            if (ln == 0) atomicAdd(&ns_dot[r0 + r + 8 * half], v);
        }
    }
    if (wavU < 4){
        int col = (wavU + 16) * 16 + ln;
        float wv = (col < Hq) ? w_ns[col] : 0.f;
#pragma unroll
        for (int r = 0; r < 8; ++r){
            float v = eluf(acc2[r]) * wv;
            for (int m = 1; m < 16; m <<= 1) v += __shfl_xor(v, m, 32);
            if (ln == 0) atomicAdd(&ns_dot[r0 + r + 8 * half], v);
        }
    }
}

// ---------------------------------------------------------------- edge WMMA kernel
// nr_raw[dst_e] += dist[src_e] * sum_k elu( sum_h instr[b,h]*edge_attrs[e,h]*We[h,k] ) * w_rs[k]
__global__ __launch_bounds__(256) void k_edge_scores(
        const float* __restrict__ edge_attrs, const float* __restrict__ instructions, int step,
        const __bf16* __restrict__ wepk, const int* __restrict__ ebat, const int* __restrict__ esrc,
        const int* __restrict__ edst, const float* __restrict__ dist,
        const float* __restrict__ w_rs, float* __restrict__ nr_dot){
    __shared__ __bf16 As[16][320];
    __shared__ int   bsh[16];
    __shared__ int   dsh[16];
    __shared__ float ssh[16];
    int tid = threadIdx.x, lane = tid & 31;
    int wavU = __builtin_amdgcn_readfirstlane(tid >> 5);
    int r0 = blockIdx.x * 16;
    if (tid < 16){
        int e = r0 + tid;
        bsh[tid] = ebat[e];
        dsh[tid] = edst[e];
        ssh[tid] = dist[esrc[e]];
        if (r0 + 16 + tid < Ee)
            __builtin_prefetch(edge_attrs + (size_t)(r0 + 16 + tid) * Hq, 0, 1);
    }
    __syncthreads();
    {   // stage scaled A tile
        int rr = tid >> 4, kb = (tid & 15) * 2;
        int b = bsh[rr];
        const float* xb = instructions + ((size_t)b * (NSTEPS + 1) + step) * Hq;
        const float* ea = edge_attrs + (size_t)(r0 + rr) * Hq;
        for (int k0 = kb; k0 < 320; k0 += 32){
            float v0 = (k0     < Hq) ? xb[k0]     * ea[k0]     : 0.f;
            float v1 = (k0 + 1 < Hq) ? xb[k0 + 1] * ea[k0 + 1] : 0.f;
            *(unsigned*)&As[rr][k0] = (unsigned)bfbits(v0) | ((unsigned)bfbits(v1) << 16);
        }
    }
    __syncthreads();
    int half = lane >> 4, ln = lane & 15;

    v8f acc0, acc1, acc2;
#pragma unroll
    for (int r = 0; r < 8; ++r){ acc0[r] = 0.f; acc1[r] = 0.f; acc2[r] = 0.f; }

    for (int kt = 0; kt < KT_PER_H; ++kt){
        const v8bf* pa = (const v8bf*)&As[ln][(kt << 5) + (half << 3)];
        v16bf a = SHUF16(pa[0], pa[2]);
        const __bf16* wb = wepk + (((size_t)kt * NT_PAD) << 9) + ((size_t)lane << 4);
        {
            const v8bf* pb = (const v8bf*)(wb + ((size_t)wavU << 9));
            v16bf bf = SHUF16(pb[0], pb[1]);
            acc0 = __builtin_amdgcn_wmma_f32_16x16x32_bf16(false, a, false, bf, (short)0, acc0, false, false);
        }
        {
            const v8bf* pb = (const v8bf*)(wb + ((size_t)(wavU + 8) << 9));
            v16bf bf = SHUF16(pb[0], pb[1]);
            acc1 = __builtin_amdgcn_wmma_f32_16x16x32_bf16(false, a, false, bf, (short)0, acc1, false, false);
        }
        if (wavU < 4){
            const v8bf* pb = (const v8bf*)(wb + ((size_t)(wavU + 16) << 9));
            v16bf bf = SHUF16(pb[0], pb[1]);
            acc2 = __builtin_amdgcn_wmma_f32_16x16x32_bf16(false, a, false, bf, (short)0, acc2, false, false);
        }
    }
    // fused epilogue: elu -> dot(w_rs) -> reduce -> scatter to destination nodes
    {
        int col = wavU * 16 + ln;
        float wv = (col < Hq) ? w_rs[col] : 0.f;
#pragma unroll
        for (int r = 0; r < 8; ++r){
            float v = eluf(acc0[r]) * wv;
            for (int m = 1; m < 16; m <<= 1) v += __shfl_xor(v, m, 32);
            if (ln == 0){ int row = r + 8 * half; atomicAdd(&nr_dot[dsh[row]], ssh[row] * v); }
        }
    }
    {
        int col = (wavU + 8) * 16 + ln;
        float wv = (col < Hq) ? w_rs[col] : 0.f;
#pragma unroll
        for (int r = 0; r < 8; ++r){
            float v = eluf(acc1[r]) * wv;
            for (int m = 1; m < 16; m <<= 1) v += __shfl_xor(v, m, 32);
            if (ln == 0){ int row = r + 8 * half; atomicAdd(&nr_dot[dsh[row]], ssh[row] * v); }
        }
    }
    if (wavU < 4){
        int col = (wavU + 16) * 16 + ln;
        float wv = (col < Hq) ? w_rs[col] : 0.f;
#pragma unroll
        for (int r = 0; r < 8; ++r){
            float v = eluf(acc2[r]) * wv;
            for (int m = 1; m < 16; m <<= 1) v += __shfl_xor(v, m, 32);
            if (ln == 0){ int row = r + 8 * half; atomicAdd(&nr_dot[dsh[row]], ssh[row] * v); }
        }
    }
}

// ---------------------------------------------------------------- segment softmax + dist update
__global__ void k_dist_update(const float* __restrict__ ns_dot, const float* __restrict__ nr_dot,
                              const int* __restrict__ nidx, const float* __restrict__ psim,
                              float* __restrict__ dist){
    int b = blockIdx.x, tid = threadIdx.x;
    __shared__ float red[256];
    float m1 = -1e30f, m2 = -1e30f;
    for (int n = tid; n < Nn; n += 256)
        if (nidx[n] == b){ m1 = fmaxf(m1, ns_dot[n]); m2 = fmaxf(m2, nr_dot[n]); }
    red[tid] = m1; __syncthreads();
    for (int s = 128; s > 0; s >>= 1){ if (tid < s) red[tid] = fmaxf(red[tid], red[tid + s]); __syncthreads(); }
    m1 = red[0]; __syncthreads();
    red[tid] = m2; __syncthreads();
    for (int s = 128; s > 0; s >>= 1){ if (tid < s) red[tid] = fmaxf(red[tid], red[tid + s]); __syncthreads(); }
    m2 = red[0]; __syncthreads();
    float s1 = 0.f, s2 = 0.f;
    for (int n = tid; n < Nn; n += 256)
        if (nidx[n] == b){ s1 += __expf(ns_dot[n] - m1); s2 += __expf(nr_dot[n] - m2); }
    red[tid] = s1; __syncthreads();
    for (int s = 128; s > 0; s >>= 1){ if (tid < s) red[tid] += red[tid + s]; __syncthreads(); }
    s1 = red[0]; __syncthreads();
    red[tid] = s2; __syncthreads();
    for (int s = 128; s > 0; s >>= 1){ if (tid < s) red[tid] += red[tid + s]; __syncthreads(); }
    s2 = red[0]; __syncthreads();
    float rs = psim[b * (Pq + 1) + Pq];
    float i1 = 1.f / s1, i2 = 1.f / s2;
    for (int n = tid; n < Nn; n += 256)
        if (nidx[n] == b){
            float nsv = __expf(ns_dot[n] - m1) * i1;
            float nrv = __expf(nr_dot[n] - m2) * i2;
            dist[n] = rs * nrv + (1.f - rs) * nsv;
        }
}

// ---------------------------------------------------------------- final aggregation + FCs
__global__ void k_agg(const float* __restrict__ node_attrs, const float* __restrict__ psim,
                      const int* __restrict__ nidx, const float* __restrict__ dist,
                      float* __restrict__ agg){
    int n = blockIdx.x;
    int b = nidx[n];
    float dn = dist[n];
    for (int h = threadIdx.x; h < Hq; h += blockDim.x){
        float s = 0.f;
#pragma unroll
        for (int p = 0; p < Pq; ++p)
            s += psim[b * (Pq + 1) + p] * node_attrs[((size_t)n * Pq + p) * Hq + h];
        atomicAdd(&agg[(size_t)b * Hq + h], dn * s);
    }
}

__global__ void k_fc1(const float* __restrict__ enc, const float* __restrict__ agg,
                      const float* __restrict__ w, const float* __restrict__ bias,
                      float* __restrict__ out){
    int i = blockIdx.x * blockDim.x + threadIdx.x;
    if (i >= Bq * 2 * Hq) return;
    int b = i / (2 * Hq), j = i % (2 * Hq);
    const float* wr = w + (size_t)j * 2 * Hq;
    const float* e  = enc + (size_t)b * Hq;
    const float* a  = agg + (size_t)b * Hq;
    float s = bias[j];
    for (int k = 0; k < Hq; ++k) s += e[k] * wr[k];
    for (int k = 0; k < Hq; ++k) s += a[k] * wr[Hq + k];
    out[i] = eluf(s);
}

__global__ void k_fc2(const float* __restrict__ x, const float* __restrict__ w,
                      const float* __restrict__ bias, float* __restrict__ out){
    int i = blockIdx.x * blockDim.x + threadIdx.x;
    if (i >= Bq * OUTq) return;
    int b = i / OUTq, o = i % OUTq;
    const float* wr = w + (size_t)o * 2 * Hq;
    const float* xr = x + (size_t)b * 2 * Hq;
    float s = bias[o];
    for (int k = 0; k < 2 * Hq; ++k) s += xr[k] * wr[k];
    out[i] = s;
}

// ---------------------------------------------------------------- workspace layout (floats)
static const size_t OFF_QK   = 0;                       // 1280*300
static const size_t OFF_SIM  = OFF_QK   + 384000;       // 1280*2049
static const size_t OFF_TAG  = OFF_SIM  + 2622720;      // 1280*300
static const size_t OFF_H    = OFF_TAG  + 384000;       // 64*300
static const size_t OFF_C    = OFF_H    + 19200;        // 64*300
static const size_t OFF_Z    = OFF_C    + 19200;        // 64*1200
static const size_t OFF_HXA  = OFF_Z    + 76800;        // 64*300
static const size_t OFF_HXB  = OFF_HXA  + 19200;        // 64*300
static const size_t OFF_HID  = OFF_HXB  + 19200;        // 64*4*300
static const size_t OFF_INS  = OFF_HID  + 76800;        // 64*4*300
static const size_t OFF_PSIM = OFF_INS  + 76800;        // 64*9
static const size_t OFF_CNT  = OFF_PSIM + 576;          // 64
static const size_t OFF_DIST = OFF_CNT  + 64;           // 32768
static const size_t OFF_NS   = OFF_DIST + 32768;        // 32768
static const size_t OFF_NR   = OFF_NS   + 32768;        // 32768  (contiguous with NS)
static const size_t OFF_AGG  = OFF_NR   + 32768;        // 64*300
static const size_t OFF_FC1  = OFF_AGG  + 19200;        // 64*600
static const size_t OFF_WEPK = OFF_FC1  + 38400;        // 10*20*512 bf16 = 51200 floats
static const size_t OFF_WPPK = OFF_WEPK + 51200;        // 8*10*20*512 bf16 = 409600 floats

extern "C" void kernel_launch(void* const* d_in, const int* in_sizes, int n_in,
                              void* d_out, int out_size, void* d_ws, size_t ws_size,
                              hipStream_t stream) {
    const float* questions = (const float*)d_in[0];
    const float* cvocab    = (const float*)d_in[1];
    const float* propemb   = (const float*)d_in[2];
    const float* node_attrs= (const float*)d_in[3];
    const float* edge_attrs= (const float*)d_in[4];
    const float* w_tag     = (const float*)d_in[5];
    const float* defe      = (const float*)d_in[6];
    const float* lWih      = (const float*)d_in[7];
    const float* lWhh      = (const float*)d_in[8];
    const float* lbih      = (const float*)d_in[9];
    const float* lbhh      = (const float*)d_in[10];
    const float* rWih      = (const float*)d_in[11];
    const float* rWhh      = (const float*)d_in[12];
    const float* rbih      = (const float*)d_in[13];
    const float* rbhh      = (const float*)d_in[14];
    const float* Wp        = (const float*)d_in[15];
    const float* We        = (const float*)d_in[16];
    const float* w_ns      = (const float*)d_in[17];
    const float* w_rs      = (const float*)d_in[18];
    const float* fc1_w     = (const float*)d_in[19];
    const float* fc1_b     = (const float*)d_in[20];
    const float* fc2_w     = (const float*)d_in[21];
    const float* fc2_b     = (const float*)d_in[22];
    const int*   nidx      = (const int*)d_in[23];
    const int*   esrc      = (const int*)d_in[24];
    const int*   edst      = (const int*)d_in[25];
    const int*   ebat      = (const int*)d_in[26];

    float* ws   = (float*)d_ws;
    float* qk   = ws + OFF_QK;
    float* sim  = ws + OFF_SIM;
    float* tag  = ws + OFF_TAG;
    float* h    = ws + OFF_H;      // LSTM hidden == "encoded" after the scan
    float* c    = ws + OFF_C;
    float* z    = ws + OFF_Z;
    float* hxA  = ws + OFF_HXA;
    float* hxB  = ws + OFF_HXB;
    float* hid  = ws + OFF_HID;
    float* ins  = ws + OFF_INS;
    float* psim = ws + OFF_PSIM;
    float* cnt  = ws + OFF_CNT;
    float* dist = ws + OFF_DIST;
    float* nsd  = ws + OFF_NS;
    float* nrd  = ws + OFF_NR;
    float* agg  = ws + OFF_AGG;
    float* fc1o = ws + OFF_FC1;
    __bf16* wepk = (__bf16*)(ws + OFF_WEPK);
    __bf16* wppk = (__bf16*)(ws + OFF_WPPK);
    float* out  = (float*)d_out;

    // ---- pre-pack GEMM weights into bf16 WMMA fragment layout (L2-resident)
    k_pack_we<<<(KT_PER_H * NT_PAD * 32 * 16 + 255) / 256, 256, 0, stream>>>(We, wepk);
    k_pack_wp<<<(Pq * KT_PER_H * NT_PAD * 32 * 16 + 255) / 256, 256, 0, stream>>>(Wp, wppk);

    // ---- tagging: sim = softmax(q @ w_tag @ vocab_ext^T); tagged mixture
    k_qk<<<(Bq * Lq * Hq + 255) / 256, 256, 0, stream>>>(questions, w_tag, qk);
    k_logits<<<(int)(((long)Bq * Lq * (Cv + 1) + 255) / 256), 256, 0, stream>>>(qk, cvocab, defe, sim);
    k_softmax_rows<<<Bq * Lq, 256, 0, stream>>>(sim, Bq * Lq, Cv + 1);
    k_tagged<<<(Bq * Lq * Hq + 255) / 256, 256, 0, stream>>>(sim, questions, cvocab, tag);

    // ---- LSTM scan over L=20 (sequential)
    k_zero<<<(2 * Bq * Hq + 255) / 256, 256, 0, stream>>>(h, 2 * Bq * Hq);   // zeros h and c
    for (int t = 0; t < Lq; ++t){
        k_lstm_z<<<(Bq * 4 * Hq + 255) / 256, 256, 0, stream>>>(tag, t, h, lWih, lWhh, lbih, lbhh, z);
        k_lstm_upd<<<(Bq * Hq + 255) / 256, 256, 0, stream>>>(z, h, c);
    }

    // ---- RNN instruction decoder (4 hiddens) + attention instructions
    k_zero<<<(Bq * Hq + 255) / 256, 256, 0, stream>>>(hxA, Bq * Hq);
    const float* hin = hxA; float* hout = hxB;
    for (int s = 0; s <= NSTEPS; ++s){
        k_rnn<<<(Bq * Hq + 255) / 256, 256, 0, stream>>>(h, hin, rWih, rWhh, rbih, rbhh, hout, hid, s);
        const float* tmp = hout; hout = (float*)hin; hin = tmp;
    }
    k_instr<<<Bq * (NSTEPS + 1), 128, 0, stream>>>(hid, tag, ins);

    // ---- graph init
    k_zero<<<1, 64, 0, stream>>>(cnt, Bq);
    k_counts<<<(Nn + 255) / 256, 256, 0, stream>>>(nidx, cnt);
    k_dist_init<<<(Nn + 255) / 256, 256, 0, stream>>>(nidx, cnt, dist);

    // ---- message passing steps: WMMA-heavy node & edge kernels, fused epilogues
    for (int step = 0; step < NSTEPS; ++step){
        k_psim<<<Bq, 32, 0, stream>>>(ins, step, propemb, psim);
        k_zero<<<(2 * Nn + 255) / 256, 256, 0, stream>>>(nsd, 2 * Nn);   // zeros ns and nr
        k_node_scores<<<Nn / 16, 256, 0, stream>>>(node_attrs, ins, step, psim, wppk, nidx, w_ns, nsd);
        k_edge_scores<<<Ee / 16, 256, 0, stream>>>(edge_attrs, ins, step, wepk, ebat, esrc, edst, dist, w_rs, nrd);
        k_dist_update<<<Bq, 256, 0, stream>>>(nsd, nrd, nidx, psim, dist);
    }

    // ---- final aggregation + classifier
    k_psim<<<Bq, 32, 0, stream>>>(ins, NSTEPS, propemb, psim);
    k_zero<<<(Bq * Hq + 255) / 256, 256, 0, stream>>>(agg, Bq * Hq);
    k_agg<<<Nn, 256, 0, stream>>>(node_attrs, psim, nidx, dist, agg);
    k_fc1<<<(Bq * 2 * Hq + 255) / 256, 256, 0, stream>>>(h, agg, fc1_w, fc1_b, fc1o);
    k_fc2<<<(Bq * OUTq + 255) / 256, 256, 0, stream>>>(fc1o, fc2_w, fc2_b, out);
}
